// CrossModalAttentionFusion_54820962566443
// MI455X (gfx1250) — matrix operand order repaired and checked by reading, not compiled, not run on hardware
//
#include <hip/hip_runtime.h>
#include <hip/hip_bf16.h>

// ---------------------------------------------------------------------------
// Problem constants (from the reference)
// ---------------------------------------------------------------------------
#define BATCH 16
#define CH    512
#define CQ    64
#define HW    1024   // 32*32
#define EPS   1e-5f

typedef __attribute__((ext_vector_type(16))) __bf16 v16bf;
typedef __attribute__((ext_vector_type(8)))  float  v8f;
typedef unsigned short u16;

// Output modes for the generic GEMM
#define OM_F32    0   // OutF[m*ldo + n] = v            (f32, row-major)
#define OM_BF16N  1   // OutH[m*ldo + n] = bf16(v)      (bf16, row-major)
#define OM_BF16T  2   // OutH[n*ldo + m] = bf16(v)      (bf16, transposed, packed 16B)

// Tensor Data Mover availability (5-arg form on ROCm7.2/clang-22, 6-arg on clang-23)
#if defined(__has_builtin)
#  if __has_builtin(__builtin_amdgcn_tensor_load_to_lds) && \
      __has_builtin(__builtin_amdgcn_s_wait_tensorcnt)
#    define HAVE_TDM 1
#  endif
#endif
#ifndef HAVE_TDM
#  define HAVE_TDM 0
#endif

// ---------------------------------------------------------------------------
// bf16 helpers (bits stored as unsigned short)
// ---------------------------------------------------------------------------
static __device__ __forceinline__ unsigned short f32_to_bf16_bits(float f) {
    union { float f; unsigned int u; } cv; cv.f = f;
    unsigned int lsb = (cv.u >> 16) & 1u;
    cv.u += 0x7FFFu + lsb;               // round-to-nearest-even
    return (unsigned short)(cv.u >> 16);
}

// ---------------------------------------------------------------------------
// f32 -> bf16 conversion kernels
// ---------------------------------------------------------------------------
__global__ void cvt_f32_bf16(const float* __restrict__ in,
                             u16* __restrict__ out, int n) {
    int i = blockIdx.x * blockDim.x + threadIdx.x;
    if (i < n) out[i] = f32_to_bf16_bits(in[i]);
}

// (B, C, HW) f32  ->  (B, HW, C) bf16   (makes GEMM B-operands n-major)
__global__ void cvt_f32_bf16_transpose(const float* __restrict__ in,
                                       u16* __restrict__ out) {
    long long i = (long long)blockIdx.x * blockDim.x + threadIdx.x;
    if (i >= (long long)BATCH * CH * HW) return;
    int n = (int)(i % HW);
    int c = (int)((i / HW) % CH);
    int b = (int)(i / ((long long)CH * HW));
    out[((long long)b * HW + n) * CH + c] = f32_to_bf16_bits(in[i]);
}

#if HAVE_TDM
// ---------------------------------------------------------------------------
// TDM 2-D tile load: tile (tileN rows x tileK elems, row stride strideElems,
// 2-byte elements) from global -> linear LDS at lds_off. Issued per-wave.
// D# layout per CDNA5 ISA section 8 (groups 0/1; groups 2/3 zero for 2-D).
// ---------------------------------------------------------------------------
typedef unsigned int u32x4 __attribute__((ext_vector_type(4)));
typedef int          i32x4 __attribute__((ext_vector_type(4)));
typedef int          i32x8 __attribute__((ext_vector_type(8)));

static __device__ __forceinline__ void tdm_load_2d(unsigned lds_off,
                                                   const u16* gsrc,
                                                   int tileK, int tileN,
                                                   int strideElems)
{
    unsigned long long ga = (unsigned long long)gsrc;
    u32x4 g0;
    g0[0] = 1u;                                            // count=1 (valid D#)
    g0[1] = lds_off;                                       // lds_addr (bytes)
    g0[2] = (unsigned)(ga & 0xffffffffull);                // global_addr[31:0]
    g0[3] = (unsigned)((ga >> 32) & 0x01ffffffull)         // global_addr[56:32]
          | (2u << 30);                                    // type=2 ("image")
    unsigned tk = (unsigned)tileK, tn = (unsigned)tileN;
    unsigned st = (unsigned)strideElems;
    i32x8 g1;
    g1[0] = (int)(1u << 16);                               // data_size=1 (2 bytes)
    g1[1] = (int)(tk << 16);                               // tensor_dim0[15:0] @d1[31:16]
    g1[2] = (int)((tk >> 16) | (tn << 16));                // dim0[31:16] | dim1[15:0]
    g1[3] = (int)((tn >> 16) | (tk << 16));                // dim1[31:16] | tile_dim0
    g1[4] = (int)(tn & 0xffffu);                           // tile_dim1 (tile_dim2=0)
    g1[5] = (int)st;                                       // tensor_dim0_stride[31:0]
    g1[6] = 0;                                             // stride0[47:32] | stride1 lo
    g1[7] = 0;
    i32x4 z4 = {0, 0, 0, 0};
#if __clang_major__ >= 23
    i32x8 z8 = {0, 0, 0, 0, 0, 0, 0, 0};
    __builtin_amdgcn_tensor_load_to_lds(g0, g1, z4, z4, z8, 0);
#else
    __builtin_amdgcn_tensor_load_to_lds(g0, g1, z4, z4, 0);
#endif
}

static __device__ __forceinline__ unsigned lds_offset_of(const void* p) {
    return (unsigned)(unsigned long long)p;   // low 32 bits = LDS byte offset
}
#endif // HAVE_TDM

// ---------------------------------------------------------------------------
// Generic batched bf16 WMMA GEMM:  Out[b] = alpha * A[b] * B[b] + bias
//   A: row-major (M x K):  A[m*lda + k]
//   B: n-major  (N x K):   B[n*ldb + k]
// Block: 256 threads = 8 waves (wave32). Tile 64(M) x 128(N), K-step 32.
// LDS tiles are LINEAR row-major (rows x 32 k-elems):
//   A fragment (lane m, half h): two 16B slices  k=8h..8h+7  and  k=16+8h..16+8h+7
//   B fragment (lane n, half h): two 16B slices  k=8h..8h+7  and  k=16+8h..16+8h+7
// TDM path: wave0 DMAs next tile into the other LDS buffer (double-buffered)
// while all waves run WMMA; sync = s_wait_tensorcnt + workgroup barrier.
// Fallback: register-pipelined global->LDS copy overlapping the WMMA block.
// Requires M%64==0, N%128==0, K%32==0, lda/ldb multiples of 8.
// ---------------------------------------------------------------------------
#define BM 64
#define BN 128
#define BK 32

template <int OM>
__global__ void __launch_bounds__(256)
gemm_bf16_wmma(const u16* __restrict__ A, long long sAb, int lda,
               const u16* __restrict__ Bsrc, long long sBb, int ldb,
               float* __restrict__ OutF, u16* __restrict__ OutH,
               long long sOb, int ldo,
               const float* __restrict__ bias, float alpha, int K)
{
#if HAVE_TDM
    __shared__ __align__(16) u16 lAf[2][BM * BK];   // 2 x 4 KB
    __shared__ __align__(16) u16 lBf[2][BN * BK];   // 2 x 8 KB
#else
    __shared__ __align__(16) u16 lAf[1][BM * BK];
    __shared__ __align__(16) u16 lBf[1][BN * BK];
#endif

    const int b  = blockIdx.z;
    const int m0 = blockIdx.y * BM;
    const int n0 = blockIdx.x * BN;
    A    += (long long)b * sAb;
    Bsrc += (long long)b * sBb;

    const int tid  = threadIdx.x;
    const int lane = tid & 31;
    const int wv   = tid >> 5;
    const int wm   = wv & 3;       // 16-row subtile
    const int wn   = wv >> 2;      // 64-col half

    v8f acc[4] = {{}, {}, {}, {}};
    const int nK = K / BK;

#if HAVE_TDM
    if (wv == 0) {
        tdm_load_2d(lds_offset_of(&lAf[0][0]), A    + (long long)m0 * lda, BK, BM, lda);
        tdm_load_2d(lds_offset_of(&lBf[0][0]), Bsrc + (long long)n0 * ldb, BK, BN, ldb);
    }
#else
    // staging: one 8-elem (16B) chunk of A, two of B, per thread
    const int sm  = tid >> 2;             // A row 0..63
    const int skc = (tid & 3) << 3;       // k-chunk {0,8,16,24}
    const u16* gA  = A    + (long long)(m0 + sm) * lda + skc;
    const u16* gB0 = Bsrc + (long long)(n0 + sm) * ldb + skc;   // B rows 0..63
    const u16* gB1 = gB0 + (long long)64 * ldb;                 // B rows 64..127
    uint4 ra  = *reinterpret_cast<const uint4*>(gA);
    uint4 rb0 = *reinterpret_cast<const uint4*>(gB0);
    uint4 rb1 = *reinterpret_cast<const uint4*>(gB1);
#endif

    for (int it = 0; it < nK; ++it) {
#if HAVE_TDM
        const int buf = it & 1;
        if (wv == 0) {
            if (it + 1 < nK) {
                const long long kn = (long long)(it + 1) * BK;
                tdm_load_2d(lds_offset_of(&lAf[buf ^ 1][0]),
                            A + (long long)m0 * lda + kn, BK, BM, lda);
                tdm_load_2d(lds_offset_of(&lBf[buf ^ 1][0]),
                            Bsrc + (long long)n0 * ldb + kn, BK, BN, ldb);
                __builtin_amdgcn_s_wait_tensorcnt((short)2);  // current tile done
            } else {
                __builtin_amdgcn_s_wait_tensorcnt((short)0);
            }
        }
        __syncthreads();
        const u16* tA = lAf[buf];
        const u16* tB = lBf[buf];
#else
        *reinterpret_cast<uint4*>(&lAf[0][sm * BK + skc])        = ra;
        *reinterpret_cast<uint4*>(&lBf[0][sm * BK + skc])        = rb0;
        *reinterpret_cast<uint4*>(&lBf[0][(sm + 64) * BK + skc]) = rb1;
        __syncthreads();
        if (it + 1 < nK) {      // issue next-tile loads; they overlap the WMMAs
            gA += BK; gB0 += BK; gB1 += BK;
            ra  = *reinterpret_cast<const uint4*>(gA);
            rb0 = *reinterpret_cast<const uint4*>(gB0);
            rb1 = *reinterpret_cast<const uint4*>(gB1);
            __builtin_prefetch(gA + BK, 0, 3);
        }
        const u16* tA = lAf[0];
        const u16* tB = lBf[0];
#endif
        // ---- fragments from linear LDS ----
        const int am = wm * 16 + (lane & 15);
        const int hh = lane >> 4;
        union { uint4 q[2]; v16bf v; } au;
        au.q[0] = *reinterpret_cast<const uint4*>(&tA[am * BK + 8 * hh]);       // k=8h..8h+7
        au.q[1] = *reinterpret_cast<const uint4*>(&tA[am * BK + 16 + 8 * hh]);  // k=16+8h..
        v16bf af = au.v;
        #pragma unroll
        for (int t = 0; t < 4; ++t) {
            int n = wn * 64 + t * 16 + (lane & 15);
            union { uint4 q[2]; v16bf v; } bu;
            bu.q[0] = *reinterpret_cast<const uint4*>(&tB[n * BK + 8 * hh]);       // k=8h..8h+7
            bu.q[1] = *reinterpret_cast<const uint4*>(&tB[n * BK + 16 + 8 * hh]);  // k=16+8h..
            v16bf bf = bu.v;
            acc[t] = __builtin_amdgcn_wmma_f32_16x16x32_bf16(
                         false, af, false, bf, (short)0, acc[t], false, false);
        }
        __syncthreads();
    }

    // ---- store D (C/D layout: VGPR r -> row r + 8*(lane>=16), col lane&15) ----
    const int cn  = lane & 15;
    const int chh = lane >> 4;
    const int gm8 = m0 + wm * 16 + 8 * chh;   // base row of this lane's 8 accs
    const long long ob = (long long)b * sOb;

    if (OM == OM_BF16T) {
        #pragma unroll
        for (int t = 0; t < 4; ++t) {
            int gn = n0 + wn * 64 + t * 16 + cn;
            union { unsigned short s[8]; uint4 v; } pk;
            #pragma unroll
            for (int r = 0; r < 8; ++r) {
                float v = acc[t][r] * alpha + (bias ? bias[gm8 + r] : 0.0f);
                pk.s[r] = f32_to_bf16_bits(v);
            }
            *reinterpret_cast<uint4*>(&OutH[ob + (long long)gn * ldo + gm8]) = pk.v;
        }
    } else {
        #pragma unroll
        for (int t = 0; t < 4; ++t) {
            int gn = n0 + wn * 64 + t * 16 + cn;
            #pragma unroll
            for (int r = 0; r < 8; ++r) {
                int gm = gm8 + r;
                float v = acc[t][r] * alpha + (bias ? bias[gm] : 0.0f);
                long long idx = ob + (long long)gm * ldo + gn;
                if (OM == OM_F32) OutF[idx] = v;
                else              OutH[idx] = f32_to_bf16_bits(v);
            }
        }
    }
}

// ---------------------------------------------------------------------------
// Row softmax: S (rows x 1024 f32) -> P (bf16 bits). One block per row.
// ---------------------------------------------------------------------------
__global__ void __launch_bounds__(256)
softmax_rows(const float* __restrict__ S, u16* __restrict__ P, int ncols)
{
    __shared__ float red[256];
    const long long row = blockIdx.x;
    const float* s = S + row * ncols;
    u16* p = P + row * ncols;
    const int tid = threadIdx.x;

    float mx = -3.4e38f;
    for (int j = tid; j < ncols; j += 256) mx = fmaxf(mx, s[j]);
    red[tid] = mx; __syncthreads();
    for (int off = 128; off > 0; off >>= 1) {
        if (tid < off) red[tid] = fmaxf(red[tid], red[tid + off]);
        __syncthreads();
    }
    mx = red[0]; __syncthreads();

    float sum = 0.f;
    for (int j = tid; j < ncols; j += 256) sum += __expf(s[j] - mx);
    red[tid] = sum; __syncthreads();
    for (int off = 128; off > 0; off >>= 1) {
        if (tid < off) red[tid] += red[tid + off];
        __syncthreads();
    }
    const float inv = 1.0f / red[0];
    for (int j = tid; j < ncols; j += 256)
        p[j] = f32_to_bf16_bits(__expf(s[j] - mx) * inv);
}

// ---------------------------------------------------------------------------
// BatchNorm statistics: one block per channel, reduce over (B, HW).
// ---------------------------------------------------------------------------
__global__ void __launch_bounds__(256)
bn_stats(const float* __restrict__ fused, float* __restrict__ mean,
         float* __restrict__ rstd)
{
    __shared__ float rs[256], rq[256];
    const int c = blockIdx.x;
    const int tid = threadIdx.x;
    float s = 0.f, q = 0.f;
    for (int b = 0; b < BATCH; ++b) {
        const float* p = fused + ((long long)b * CH + c) * HW;
        for (int j = tid; j < HW; j += 256) { float v = p[j]; s += v; q += v * v; }
    }
    rs[tid] = s; rq[tid] = q; __syncthreads();
    for (int off = 128; off > 0; off >>= 1) {
        if (tid < off) { rs[tid] += rs[tid + off]; rq[tid] += rq[tid + off]; }
        __syncthreads();
    }
    if (tid == 0) {
        const float n = (float)(BATCH * HW);
        float m   = rs[0] / n;
        float var = rq[0] / n - m * m;     // biased variance (matches jnp.var)
        mean[c] = m;
        rstd[c] = rsqrtf(var + EPS);
    }
}

// ---------------------------------------------------------------------------
// BN apply + affine + ReLU
// ---------------------------------------------------------------------------
__global__ void bn_apply_relu(const float* __restrict__ fused,
                              const float* __restrict__ mean,
                              const float* __restrict__ rstd,
                              const float* __restrict__ gamma,
                              const float* __restrict__ beta,
                              float* __restrict__ out, long long total)
{
    long long i = (long long)blockIdx.x * blockDim.x + threadIdx.x;
    if (i >= total) return;
    int c = (int)((i / HW) % CH);
    float v = (fused[i] - mean[c]) * rstd[c] * gamma[c] + beta[c];
    out[i] = fmaxf(v, 0.0f);
}

// ---------------------------------------------------------------------------
// Host-side orchestration
// ---------------------------------------------------------------------------
extern "C" void kernel_launch(void* const* d_in, const int* in_sizes, int n_in,
                              void* d_out, int out_size, void* d_ws, size_t ws_size,
                              hipStream_t stream) {
    (void)in_sizes; (void)n_in; (void)out_size; (void)ws_size;

    const float* rgb    = (const float*)d_in[0];
    const float* depth  = (const float*)d_in[1];
    const float* Wq_rgb = (const float*)d_in[2];
    const float* bq_rgb = (const float*)d_in[3];
    const float* Wk_dep = (const float*)d_in[4];
    const float* bk_dep = (const float*)d_in[5];
    const float* Wv_dep = (const float*)d_in[6];
    const float* bv_dep = (const float*)d_in[7];
    const float* Wq_dep = (const float*)d_in[8];
    const float* bq_dep = (const float*)d_in[9];
    const float* Wk_rgb = (const float*)d_in[10];
    const float* bk_rgb = (const float*)d_in[11];
    const float* Wv_rgb = (const float*)d_in[12];
    const float* bv_rgb = (const float*)d_in[13];
    const float* W_fuse = (const float*)d_in[14];
    const float* gamma  = (const float*)d_in[15];
    const float* beta   = (const float*)d_in[16];
    float* out = (float*)d_out;

    // ---- workspace carve-up (256B aligned) ----
    char* base = (char*)d_ws;
    size_t off = 0;
    auto alloc = [&](size_t bytes) -> char* {
        char* p = base + off;
        off = (off + bytes + 255) & ~(size_t)255;
        return p;
    };
    u16*  rgbT      = (u16*)alloc((size_t)BATCH * HW * CH * 2);   // (B,HW,C) bf16
    u16*  depT      = (u16*)alloc((size_t)BATCH * HW * CH * 2);
    u16*  wq_rgb_bf = (u16*)alloc((size_t)CQ * CH * 2);
    u16*  wk_dep_bf = (u16*)alloc((size_t)CQ * CH * 2);
    u16*  wv_dep_bf = (u16*)alloc((size_t)CH * CH * 2);
    u16*  wq_dep_bf = (u16*)alloc((size_t)CQ * CH * 2);
    u16*  wk_rgb_bf = (u16*)alloc((size_t)CQ * CH * 2);
    u16*  wv_rgb_bf = (u16*)alloc((size_t)CH * CH * 2);
    u16*  wfuse_bf  = (u16*)alloc((size_t)CH * 2 * CH * 2);
    u16*  qT_rgb    = (u16*)alloc((size_t)BATCH * HW * CQ * 2);   // (B,HW,CQ)
    u16*  kT_dep    = (u16*)alloc((size_t)BATCH * HW * CQ * 2);
    u16*  qT_dep    = (u16*)alloc((size_t)BATCH * HW * CQ * 2);
    u16*  kT_rgb    = (u16*)alloc((size_t)BATCH * HW * CQ * 2);
    u16*  v_dep     = (u16*)alloc((size_t)BATCH * CH * HW * 2);   // (B,C,HW)
    u16*  v_rgb     = (u16*)alloc((size_t)BATCH * CH * HW * 2);
    float* Sbuf     = (float*)alloc((size_t)BATCH * HW * HW * 4); // scores f32
    u16*  Pbuf      = (u16*)alloc((size_t)BATCH * HW * HW * 2);   // probs (n,m)
    u16*  catT      = (u16*)alloc((size_t)BATCH * HW * 2 * CH * 2); // (B,HW,2C)
    float* fused    = (float*)alloc((size_t)BATCH * CH * HW * 4);
    float* meanb    = (float*)alloc((size_t)CH * 4);
    float* rstdb    = (float*)alloc((size_t)CH * 4);

    const float inv_scale = 0.044194173824159216f;  // 1/sqrt(512)
    const long long nBCHW = (long long)BATCH * CH * HW;

    // ---- 1) conversions ----
    cvt_f32_bf16_transpose<<<(unsigned)((nBCHW + 255) / 256), 256, 0, stream>>>(rgb, rgbT);
    cvt_f32_bf16_transpose<<<(unsigned)((nBCHW + 255) / 256), 256, 0, stream>>>(depth, depT);
    auto cvt = [&](const float* src, u16* dst, int n) {
        cvt_f32_bf16<<<(n + 255) / 256, 256, 0, stream>>>(src, dst, n);
    };
    cvt(Wq_rgb, wq_rgb_bf, CQ * CH);
    cvt(Wk_dep, wk_dep_bf, CQ * CH);
    cvt(Wv_dep, wv_dep_bf, CH * CH);
    cvt(Wq_dep, wq_dep_bf, CQ * CH);
    cvt(Wk_rgb, wk_rgb_bf, CQ * CH);
    cvt(Wv_rgb, wv_rgb_bf, CH * CH);
    cvt(W_fuse, wfuse_bf,  CH * 2 * CH);

    // ---- 2) projections ----
    auto proj_qk = [&](const u16* W, const float* bias, const u16* XT, u16* YT) {
        dim3 grid(HW / BN, CQ / BM, BATCH);
        gemm_bf16_wmma<OM_BF16T><<<grid, 256, 0, stream>>>(
            W, 0, CH,
            XT, (long long)HW * CH, CH,
            nullptr, YT, (long long)HW * CQ, CQ,
            bias, 1.0f, CH);
    };
    auto proj_v = [&](const u16* W, const float* bias, const u16* XT, u16* Y) {
        dim3 grid(HW / BN, CH / BM, BATCH);
        gemm_bf16_wmma<OM_BF16N><<<grid, 256, 0, stream>>>(
            W, 0, CH,
            XT, (long long)HW * CH, CH,
            nullptr, Y, (long long)CH * HW, HW,
            bias, 1.0f, CH);
    };
    proj_qk(wq_rgb_bf, bq_rgb, rgbT, qT_rgb);
    proj_qk(wk_dep_bf, bk_dep, depT, kT_dep);
    proj_v (wv_dep_bf, bv_dep, depT, v_dep);
    proj_qk(wq_dep_bf, bq_dep, depT, qT_dep);
    proj_qk(wk_rgb_bf, bk_rgb, rgbT, kT_rgb);
    proj_v (wv_rgb_bf, bv_rgb, rgbT, v_rgb);

    // ---- 3) attention (each direction); outputs go into catT halves ----
    auto attention = [&](const u16* QT, const u16* KT, const u16* V, u16* catT_col) {
        dim3 gs(HW / BN, HW / BM, BATCH);
        gemm_bf16_wmma<OM_F32><<<gs, 256, 0, stream>>>(
            QT, (long long)HW * CQ, CQ,
            KT, (long long)HW * CQ, CQ,
            Sbuf, nullptr, (long long)HW * HW, HW,
            nullptr, inv_scale, CQ);
        softmax_rows<<<BATCH * HW, 256, 0, stream>>>(Sbuf, Pbuf, HW);
        dim3 ga(HW / BN, CH / BM, BATCH);
        gemm_bf16_wmma<OM_BF16T><<<ga, 256, 0, stream>>>(
            V, (long long)CH * HW, HW,
            Pbuf, (long long)HW * HW, HW,
            nullptr, catT_col, (long long)HW * 2 * CH, 2 * CH,
            nullptr, 1.0f, HW);
    };
    attention(qT_rgb, kT_dep, v_dep, catT);        // rgb_att -> catT cols [0,512)
    attention(qT_dep, kT_rgb, v_rgb, catT + CH);   // dep_att -> catT cols [512,1024)

    // ---- 4) fuse: fused[b](o,n) = sum_c2 Wf[o,c2] * catT[n,c2] ----
    {
        dim3 gf(HW / BN, CH / BM, BATCH);
        gemm_bf16_wmma<OM_F32><<<gf, 256, 0, stream>>>(
            wfuse_bf, 0, 2 * CH,
            catT, (long long)HW * 2 * CH, 2 * CH,
            fused, nullptr, (long long)CH * HW, HW,
            nullptr, 1.0f, 2 * CH);
    }

    // ---- 5) batchnorm (batch stats) + affine + relu ----
    bn_stats<<<CH, 256, 0, stream>>>(fused, meanb, rstdb);
    bn_apply_relu<<<(unsigned)((nBCHW + 255) / 256), 256, 0, stream>>>(
        fused, meanb, rstdb, gamma, beta, out, nBCHW);
}
